// KuramotoInspiredLayer_39754217292302
// MI455X (gfx1250) — matrix operand order repaired and checked by reading, not compiled, not run on hardware
//
#include <hip/hip_runtime.h>
#include <hip/hip_fp16.h>

typedef __attribute__((ext_vector_type(16))) _Float16 v16h;
typedef __attribute__((ext_vector_type(8)))  _Float16 v8h;
typedef __attribute__((ext_vector_type(8)))  float    v8f;

#define WMMA_F16(a, b, c) \
  __builtin_amdgcn_wmma_f32_16x16x32_f16(false, (a), false, (b), (short)0, (c), false, false)

// ---------------------------------------------------------------------------
// Fragment loaders.
// A-fragment (16x32 f16, ISA 7.12.2): lane l (row m=l&15) holds K runs
// [kb, kb+8) and [kb+16, kb+24) where kb = 32*ks + (l<16 ? 0 : 8).
// B-fragment (32x16 f16): lane l (col n=l&15) holds K runs
// [kb, kb+8) and [kb+16, kb+24) where kb = 32*ks + (l<16 ? 0 : 8); weights
// are stored transposed (Wt[n][k]) so these are contiguous runs.
// ---------------------------------------------------------------------------
__device__ __forceinline__ v16h a_frag_from_f16(const _Float16* __restrict__ p) {
  v8h lo = *(const v8h*)p;          // 16B load
  v8h hi = *(const v8h*)(p + 16);   // 16B load
  v16h a;
#pragma unroll
  for (int i = 0; i < 8; ++i) { a[i] = lo[i]; a[8 + i] = hi[i]; }
  return a;
}

__device__ __forceinline__ v16h a_frag_from_f32_sum(const float* __restrict__ p,
                                                    const float* __restrict__ q) {
  v16h a;
#pragma unroll
  for (int i = 0; i < 8; ++i) a[i] = (_Float16)(p[i] + q[i]);
#pragma unroll
  for (int i = 0; i < 8; ++i) a[8 + i] = (_Float16)(p[16 + i] + q[16 + i]);
  return a;
}

__device__ __forceinline__ v16h b_frag_from_lds(const _Float16* p) {
  v8h lo = *(const v8h*)p;
  v8h hi = *(const v8h*)(p + 16);
  v16h b;
#pragma unroll
  for (int i = 0; i < 8; ++i) { b[i] = lo[i]; b[8 + i] = hi[i]; }
  return b;
}

__device__ __forceinline__ void atomic_add_f32(float* p, float v) {
  unsafeAtomicAdd(p, v);  // global_atomic_add_f32, no return
}

// ---------------------------------------------------------------------------
// Prep kernels
// ---------------------------------------------------------------------------
__global__ void prep_weight_kernel(const float* __restrict__ w,
                                   _Float16* __restrict__ wt, int K, int N) {
  int i = blockIdx.x * blockDim.x + threadIdx.x;
  if (i < K * N) {
    int k = i / N, n = i - k * N;
    wt[n * K + k] = (_Float16)w[i];  // transpose + downconvert
  }
}

// x (f32) -> xh (f16) once; also zero the aggregation buffer in the same pass
__global__ void cvt_x_kernel(const float* __restrict__ x,
                             _Float16* __restrict__ xh,
                             float* __restrict__ agg, int n) {
  int i = blockIdx.x * blockDim.x + threadIdx.x;
  if (i < n) {
    xh[i] = (_Float16)x[i];
    agg[i] = 0.0f;
  }
}

// ---------------------------------------------------------------------------
// Fused 2-layer MLP over node rows: out = act(in @ W1 + b1) @ W2 + b2
// One 16-row tile per wave; K=64 via two 16x16x32 WMMAs per 16-col tile.
// ACT: 1 = tanh, 0 = relu.  SUM: input = xa + xb (f32), else input = xh (f16).
// ---------------------------------------------------------------------------
template <int ACT, bool SUM>
__global__ __launch_bounds__(128) void node_mlp_kernel(
    const _Float16* __restrict__ xh,
    const float* __restrict__ xa, const float* __restrict__ xb,
    const _Float16* __restrict__ w1t, const float* __restrict__ b1,
    const _Float16* __restrict__ w2t, const float* __restrict__ b2,
    float* __restrict__ out, int nrows) {
  __shared__ __align__(16) _Float16 sW1[64 * 64];
  __shared__ __align__(16) _Float16 sW2[64 * 64];
  __shared__ __align__(16) _Float16 sHid[4][16 * 64];

  for (int i = threadIdx.x; i < 64 * 64 / 8; i += blockDim.x) {
    ((v8h*)sW1)[i] = ((const v8h*)w1t)[i];
    ((v8h*)sW2)[i] = ((const v8h*)w2t)[i];
  }
  __syncthreads();

  const int wave = threadIdx.x >> 5;
  const int lane = threadIdx.x & 31;
  const int l16  = lane & 15;
  const int hi   = lane >> 4;
  const int ntiles = (nrows + 15) >> 4;
  _Float16* hid = sHid[wave];

  for (int tile = blockIdx.x * 4 + wave; tile < ntiles; tile += gridDim.x * 4) {
    const int row0 = tile << 4;
    int m = row0 + l16;
    if (m >= nrows) m = nrows - 1;
    const size_t roff = (size_t)m * 64 + hi * 8;

    v16h a0, a1;
    if (SUM) {
      a0 = a_frag_from_f32_sum(xa + roff, xb + roff);
      a1 = a_frag_from_f32_sum(xa + roff + 32, xb + roff + 32);
    } else {
      a0 = a_frag_from_f16(xh + roff);
      a1 = a_frag_from_f16(xh + roff + 32);
    }

    v8f acc[4] = {};
#pragma unroll
    for (int nt = 0; nt < 4; ++nt) {
      const _Float16* wr = sW1 + (nt * 16 + l16) * 64 + hi * 8;
      acc[nt] = WMMA_F16(a0, b_frag_from_lds(wr), acc[nt]);
      acc[nt] = WMMA_F16(a1, b_frag_from_lds(wr + 32), acc[nt]);
    }

    // bias + activation, relayout C->A through per-wave LDS tile
#pragma unroll
    for (int nt = 0; nt < 4; ++nt) {
      float bb = b1[nt * 16 + l16];
#pragma unroll
      for (int r = 0; r < 8; ++r) {
        float v = acc[nt][r] + bb;
        v = ACT ? tanhf(v) : fmaxf(v, 0.0f);
        hid[(r + hi * 8) * 64 + nt * 16 + l16] = (_Float16)v;
      }
    }
    // intra-wave LDS ops are in-order: no barrier needed (wave owns its tile)
    v16h h0 = a_frag_from_f16(hid + l16 * 64 + hi * 8);
    v16h h1 = a_frag_from_f16(hid + l16 * 64 + 32 + hi * 8);

    v8f o[4] = {};
#pragma unroll
    for (int nt = 0; nt < 4; ++nt) {
      const _Float16* wr = sW2 + (nt * 16 + l16) * 64 + hi * 8;
      o[nt] = WMMA_F16(h0, b_frag_from_lds(wr), o[nt]);
      o[nt] = WMMA_F16(h1, b_frag_from_lds(wr + 32), o[nt]);
    }
#pragma unroll
    for (int nt = 0; nt < 4; ++nt) {
      float bb = b2[nt * 16 + l16];
#pragma unroll
      for (int r = 0; r < 8; ++r) {
        int rr = row0 + r + hi * 8;
        if (rr < nrows) out[(size_t)rr * 64 + nt * 16 + l16] = o[nt][r] + bb;
      }
    }
  }
}

// ---------------------------------------------------------------------------
// Edge kernel: 16 edges per wave.
// ef = [xh[tgt], xh[src]] (16x128 f16 A, gathered straight as f16),
// h = relu(ef @ cw1 + cb1) via 16 WMMAs,
// k = sigmoid(h . cw2 + cb2) via intra-half shfl_xor reduction,
// msg = k * sin(phases[src]-phases[tgt]) scattered with f32 atomics.
// ---------------------------------------------------------------------------
__global__ __launch_bounds__(128) void edge_kernel(
    const _Float16* __restrict__ xh, const long long* __restrict__ eidx, int nedges,
    const _Float16* __restrict__ cw1t, const float* __restrict__ cb1,
    const float* __restrict__ cw2, const float* __restrict__ cb2,
    const float* __restrict__ phases, float* __restrict__ agg) {
  __shared__ __align__(16) _Float16 sW[64 * 128];
  __shared__ long long sS[4][16];
  __shared__ long long sT[4][16];
  __shared__ float sK[4][16];

  for (int i = threadIdx.x; i < 64 * 128 / 8; i += blockDim.x)
    ((v8h*)sW)[i] = ((const v8h*)cw1t)[i];
  __syncthreads();

  const int wave = threadIdx.x >> 5;
  const int lane = threadIdx.x & 31;
  const int l16  = lane & 15;
  const int hi   = lane >> 4;
  const int ntiles = nedges >> 4;

  for (int tile = blockIdx.x * 4 + wave; tile < ntiles; tile += gridDim.x * 4) {
    const int e = (tile << 4) + l16;
    const long long s = eidx[e];
    const long long t = eidx[nedges + e];
    if (hi == 0) { sS[wave][l16] = s; sT[wave][l16] = t; }

    // warm up phase rows while the GEMM runs (global_prefetch_b8)
    __builtin_prefetch(phases + (size_t)s * 64, 0, 3);
    __builtin_prefetch(phases + (size_t)t * 64, 0, 3);

    const _Float16* pt = xh + (size_t)t * 64;
    const _Float16* ps = xh + (size_t)s * 64;
    v16h a[4];
#pragma unroll
    for (int ks = 0; ks < 4; ++ks) {
      int kb = ks * 32 + hi * 8;  // runs [kb,kb+8) and [kb+16,kb+24) stay in one half
      const _Float16* p = (kb < 64) ? (pt + kb) : (ps + (kb - 64));
      a[ks] = a_frag_from_f16(p);
    }

    v8f acc[4] = {};
#pragma unroll
    for (int nt = 0; nt < 4; ++nt) {
      const _Float16* wr = sW + (nt * 16 + l16) * 128 + hi * 8;
#pragma unroll
      for (int ks = 0; ks < 4; ++ks)
        acc[nt] = WMMA_F16(a[ks], b_frag_from_lds(wr + ks * 32), acc[nt]);
    }

    // relu(h) . cw2, reduced across each 16-lane half (rows r / r+8)
    float part[8];
#pragma unroll
    for (int r = 0; r < 8; ++r) part[r] = 0.0f;
#pragma unroll
    for (int nt = 0; nt < 4; ++nt) {
      float w  = cw2[nt * 16 + l16];
      float bb = cb1[nt * 16 + l16];
#pragma unroll
      for (int r = 0; r < 8; ++r)
        part[r] += fmaxf(acc[nt][r] + bb, 0.0f) * w;
    }
#pragma unroll
    for (int off = 1; off < 16; off <<= 1) {
#pragma unroll
      for (int r = 0; r < 8; ++r) part[r] += __shfl_xor(part[r], off, 32);
    }
    if (l16 == 0) {
      float cb = cb2[0];
#pragma unroll
      for (int r = 0; r < 8; ++r)
        sK[wave][hi * 8 + r] = 1.0f / (1.0f + __expf(-(part[r] + cb)));
    }

    // messages: lanes cover the 64 components in two 32-wide strips
#pragma unroll 4
    for (int e2 = 0; e2 < 16; ++e2) {
      const long long s2 = sS[wave][e2];
      const long long t2 = sT[wave][e2];
      const float kk = sK[wave][e2];
      const float* pps = phases + (size_t)s2 * 64;
      const float* ppt = phases + (size_t)t2 * 64;
      float* pa = agg + (size_t)t2 * 64;
#pragma unroll
      for (int c = 0; c < 2; ++c) {
        int j = lane + c * 32;
        float d = pps[j] - ppt[j];
        atomic_add_f32(pa + j, kk * __sinf(d));
      }
    }
  }
}

// ---------------------------------------------------------------------------
// Launch
// ---------------------------------------------------------------------------
extern "C" void kernel_launch(void* const* d_in, const int* in_sizes, int n_in,
                              void* d_out, int out_size, void* d_ws, size_t ws_size,
                              hipStream_t stream) {
  const float*     x    = (const float*)d_in[0];
  const long long* eidx = (const long long*)d_in[1];
  const float* pw1 = (const float*)d_in[2];  const float* pb1 = (const float*)d_in[3];
  const float* pw2 = (const float*)d_in[4];  const float* pb2 = (const float*)d_in[5];
  const float* fw1 = (const float*)d_in[6];  const float* fb1 = (const float*)d_in[7];
  const float* fw2 = (const float*)d_in[8];  const float* fb2 = (const float*)d_in[9];
  const float* cw1 = (const float*)d_in[10]; const float* cb1 = (const float*)d_in[11];
  const float* cw2 = (const float*)d_in[12]; const float* cb2 = (const float*)d_in[13];
  const float* ow1 = (const float*)d_in[14]; const float* ob1 = (const float*)d_in[15];
  const float* ow2 = (const float*)d_in[16]; const float* ob2 = (const float*)d_in[17];

  const int N = in_sizes[0] / 64;      // 100000
  const int E = in_sizes[1] / 2;       // 1600000
  const int D = 64;

  char* ws = (char*)d_ws;
  float* phases   = (float*)ws;    ws += (size_t)N * D * sizeof(float);
  float* freqs    = (float*)ws;    ws += (size_t)N * D * sizeof(float);
  float* agg      = (float*)ws;    ws += (size_t)N * D * sizeof(float);
  _Float16* xh    = (_Float16*)ws; ws += (size_t)N * D * sizeof(_Float16);
  _Float16* pw1t  = (_Float16*)ws; ws += 64 * 64 * sizeof(_Float16);
  _Float16* pw2t  = (_Float16*)ws; ws += 64 * 64 * sizeof(_Float16);
  _Float16* fw1t  = (_Float16*)ws; ws += 64 * 64 * sizeof(_Float16);
  _Float16* fw2t  = (_Float16*)ws; ws += 64 * 64 * sizeof(_Float16);
  _Float16* ow1t  = (_Float16*)ws; ws += 64 * 64 * sizeof(_Float16);
  _Float16* ow2t  = (_Float16*)ws; ws += 64 * 64 * sizeof(_Float16);
  _Float16* cw1t  = (_Float16*)ws; ws += 128 * 64 * sizeof(_Float16);

  // 1) prep: weights (transpose + f32->f16), x downconvert, agg zero
  prep_weight_kernel<<<(64 * 64 + 255) / 256, 256, 0, stream>>>(pw1, pw1t, 64, 64);
  prep_weight_kernel<<<(64 * 64 + 255) / 256, 256, 0, stream>>>(pw2, pw2t, 64, 64);
  prep_weight_kernel<<<(64 * 64 + 255) / 256, 256, 0, stream>>>(fw1, fw1t, 64, 64);
  prep_weight_kernel<<<(64 * 64 + 255) / 256, 256, 0, stream>>>(fw2, fw2t, 64, 64);
  prep_weight_kernel<<<(64 * 64 + 255) / 256, 256, 0, stream>>>(ow1, ow1t, 64, 64);
  prep_weight_kernel<<<(64 * 64 + 255) / 256, 256, 0, stream>>>(ow2, ow2t, 64, 64);
  prep_weight_kernel<<<(128 * 64 + 255) / 256, 256, 0, stream>>>(cw1, cw1t, 128, 64);
  cvt_x_kernel<<<((N * D) + 255) / 256, 256, 0, stream>>>(x, xh, agg, N * D);

  // 2) node MLPs (f16 A gathers straight from xh)
  const int tilesN = (N + 15) / 16;
  const int gridN  = (tilesN + 3) / 4;
  node_mlp_kernel<1, false><<<gridN, 128, 0, stream>>>(xh, nullptr, nullptr,
                                                       pw1t, pb1, pw2t, pb2, phases, N);
  node_mlp_kernel<0, false><<<gridN, 128, 0, stream>>>(xh, nullptr, nullptr,
                                                       fw1t, fb1, fw2t, fb2, freqs, N);

  // 3) edge MLP + gated sin messages + segment-sum via f32 atomics
  const int tilesE = E / 16;
  const int gridE  = (tilesE + 3) / 4;
  edge_kernel<<<gridE, 128, 0, stream>>>(xh, eidx, E, cw1t, cb1, cw2, cb2, phases, agg);

  // 4) output MLP on (freqs + agg), writes d_out
  node_mlp_kernel<0, true><<<gridN, 128, 0, stream>>>(nullptr, freqs, agg,
                                                      ow1t, ob1, ow2t, ob2,
                                                      (float*)d_out, N);
}
